// MACE_model_forcefield_56495999812201
// MI455X (gfx1250) — compile-verified, matrix operand order checked by reading, not run on hardware
//
#include <hip/hip_runtime.h>
#include <hip/hip_bf16.h>
#include <math.h>

#define NA 20000
#define NE 320000
#define NG 128
#define RCUT 10.0f
#define INV_AVG 0.0625f
#define PI_F 3.14159265358979f
#define S3C 1.73205080757f
#define S5C 2.23606797750f
#define S15C 3.87298334621f

constexpr int LMAP_[9] = {0,1,1,1,2,2,2,2,2};

typedef __attribute__((ext_vector_type(16))) _Float16 v16h;
typedef __attribute__((ext_vector_type(8)))  _Float16 v8h;
typedef __attribute__((ext_vector_type(8)))  float    v8f;
union AF { v16h v; v8h q[2]; _Float16 h[16]; };
union CF { v8f  v; float     f[8];  };

#define WMMA(A,B,C) __builtin_amdgcn_wmma_f32_16x16x32_f16(false,(A),false,(B),(short)0,(C),false,false)

// CDNA5 16-bit A/B fragment layout (ISA 7.12.2): lanes 0-15 hold K={0..7,16..23},
// lanes 16-31 hold K={8..15,24..31}; M (A) / N (B) = lane&15.
__device__ __forceinline__ int wk(int lane, int i){ return ((lane & 16) ? 8 : 0) + (i < 8 ? i : i + 8); }

// B-fragments are pre-swizzled in LDS to [tile][lane][16 halves] -> one aligned 32B load.
__device__ __forceinline__ v16h ldbs(const _Float16* w, int tile, int lane){
  return *reinterpret_cast<const v16h*>(w + ((tile*32 + lane) << 4));
}

// A-fragment reload from row-major [16 rows][32 K] half tile: per lane two
// contiguous 16B chunks (K = base..base+7 and base+16..base+23).
__device__ __forceinline__ void lda(AF& a, const _Float16* t, int m, int lane){
  int base = (lane & 16) ? 8 : 0;
  const _Float16* p = t + m*32 + base;
  a.q[0] = *reinterpret_cast<const v8h*>(p);
  a.q[1] = *reinterpret_cast<const v8h*>(p + 16);
}

// fast activations: result feeds fp16 WMMA, so raw v_rcp_f32 (1 ulp) is plenty.
__device__ __forceinline__ float frcp(float x){ return __builtin_amdgcn_rcpf(x); }
__device__ __forceinline__ float silu_f(float x){ return x*frcp(1.f+__expf(-x)); }
__device__ __forceinline__ float dsilu_f(float x){ float s=frcp(1.f+__expf(-x)); return s*(1.f+x*(1.f-s)); }

// ---------------------------------------------------------------- zero
__global__ void k_zero(float* __restrict__ A, float* __restrict__ out){
  int i = blockIdx.x*256 + threadIdx.x;        // grid covers NA*288 exactly
  A[i] = 0.f;
  if (i < NG + NA*3) out[i] = 0.f;
}

// ---------------------------------------------------------------- node embedding
__global__ void k_node_embed(const int* __restrict__ zn, const float* __restrict__ Wi,
                             const float* __restrict__ bi, float* __restrict__ nf){
  int i = blockIdx.x*256 + threadIdx.x;        // NA*32
  int n = i>>5, c = i&31;
  nf[i] = Wi[zn[n]*32 + c] + bi[c];
}

// ---------------------------------------------------------------- edge geometry: sph + radial basis
__global__ void k_edge_geom(const float* __restrict__ pos, const float* __restrict__ sh,
                            const int* __restrict__ eidx,
                            float* __restrict__ sph, float* __restrict__ ebuf){
  int e = blockIdx.x*256 + threadIdx.x;        // NE
  int s = eidx[e], r = eidx[NE+e];
  float vx = pos[r*3+0]-pos[s*3+0]+sh[(size_t)e*3+0];
  float vy = pos[r*3+1]-pos[s*3+1]+sh[(size_t)e*3+1];
  float vz = pos[r*3+2]-pos[s*3+2]+sh[(size_t)e*3+2];
  float len = sqrtf(vx*vx+vy*vy+vz*vz+1e-12f);
  float inv = frcp(len);
  float x = vx*inv, y = vy*inv, z = vz*inv;
  float* sp = sph + (size_t)e*9;
  sp[0]=1.f; sp[1]=S3C*x; sp[2]=S3C*y; sp[3]=S3C*z;
  sp[4]=S15C*x*y; sp[5]=S15C*y*z; sp[6]=0.5f*S5C*(3.f*z*z-1.f);
  sp[7]=S15C*x*z; sp[8]=0.5f*S15C*(x*x-y*y);
  float uu = len*(1.f/RCUT);
  float fc = 0.f;
  if (uu < 1.f){
    float u2=uu*uu, u3=u2*uu, u6=u3*u3;
    fc = 1.f - 28.f*u6 + 48.f*u6*uu - 21.f*u6*u2;
  }
  float rs = fmaxf(len, 1e-6f), irs = frcp(rs);
  float* eo = ebuf + (size_t)e*8;
#pragma unroll
  for (int n=1;n<=8;n++){
    float w = (float)n*(PI_F/RCUT);
    eo[n-1] = 0.44721360f * __sinf(w*rs)*irs * fc;   // sqrt(2/rc)=0.447214
  }
}

// ---------------------------------------------------------------- forward radial MLP (WMMA), 16 edges/wave
__global__ void __launch_bounds__(256) k_edge_mlp_fwd(
  const float* __restrict__ ebuf,
  const float* __restrict__ Wr1,const float* __restrict__ br1,
  const float* __restrict__ Wr2,const float* __restrict__ br2,
  const float* __restrict__ Wr3,const float* __restrict__ br3,
  const float* __restrict__ Wr4,const float* __restrict__ br4,
  _Float16* __restrict__ R)
{
  // weights pre-swizzled to fragment order: [tile][lane][16]
  __shared__ __align__(32) _Float16 w1[1024], w2[1024], w3[1024], w4[3072];
  __shared__ __align__(32) _Float16 xbuf[8][512];
  int tid = threadIdx.x;
  for (int i=tid;i<1024;i+=256){
    int lane=(i>>4)&31, ii=i&15;
    int K=wk(lane,ii), n=((i>>9)<<4)+(lane&15);
    w1[i] = (_Float16)((K<8)? Wr1[(K<<5)+n] : 0.f);
    w2[i] = (_Float16)Wr2[(K<<5)+n];
    w3[i] = (_Float16)Wr3[(K<<5)+n];
  }
  for (int i=tid;i<3072;i+=256){
    int lane=(i>>4)&31, ii=i&15;
    int K=wk(lane,ii), n=((i>>9)<<4)+(lane&15);
    w4[i] = (_Float16)Wr4[K*96+n];
  }
  __syncthreads();

  int wv=tid>>5, lane=tid&31;
  int ebase = (blockIdx.x*8+wv)*16;
  int m=lane&15, rofs=(lane&16)?8:0;
  _Float16* xb = xbuf[wv];

  AF a;
#pragma unroll
  for (int i=0;i<16;i++){ int K=wk(lane,i);
    a.h[i] = (K<8)? (_Float16)ebuf[(size_t)(ebase+m)*8+K] : (_Float16)0.f; }

  { CF c0={}, c1={};                                   // 8->32 (K padded)
    c0.v = WMMA(a.v, ldbs(w1,0,lane), c0.v);
    c1.v = WMMA(a.v, ldbs(w1,1,lane), c1.v);
#pragma unroll
    for (int v=0;v<8;v++){ int row=v+rofs;
      xb[row*32+m]    = (_Float16)silu_f(c0.f[v]+br1[m]);
      xb[row*32+16+m] = (_Float16)silu_f(c1.f[v]+br1[16+m]); } }
  __syncthreads();
  lda(a, xb, m, lane);
  __syncthreads();

  { CF c0={}, c1={};                                   // 32->32
    c0.v = WMMA(a.v, ldbs(w2,0,lane), c0.v);
    c1.v = WMMA(a.v, ldbs(w2,1,lane), c1.v);
#pragma unroll
    for (int v=0;v<8;v++){ int row=v+rofs;
      xb[row*32+m]    = (_Float16)silu_f(c0.f[v]+br2[m]);
      xb[row*32+16+m] = (_Float16)silu_f(c1.f[v]+br2[16+m]); } }
  __syncthreads();
  lda(a, xb, m, lane);
  __syncthreads();

  { CF c0={}, c1={};                                   // 32->32
    c0.v = WMMA(a.v, ldbs(w3,0,lane), c0.v);
    c1.v = WMMA(a.v, ldbs(w3,1,lane), c1.v);
#pragma unroll
    for (int v=0;v<8;v++){ int row=v+rofs;
      xb[row*32+m]    = (_Float16)silu_f(c0.f[v]+br3[m]);
      xb[row*32+16+m] = (_Float16)silu_f(c1.f[v]+br3[16+m]); } }
  __syncthreads();
  lda(a, xb, m, lane);
  __syncthreads();

  for (int nt=0; nt<6; nt++){                          // 32->96
    CF c={};
    c.v = WMMA(a.v, ldbs(w4,nt,lane), c.v);
    int col = nt*16+m;
#pragma unroll
    for (int v=0;v<8;v++) R[(size_t)(ebase+v+rofs)*96+col] = (_Float16)(c.f[v]+br4[col]);
  }
}

// ---------------------------------------------------------------- message scatter into A (L2-resident atomics)
__global__ void k_edge_scatter(const int* __restrict__ eidx, const float* __restrict__ nf,
                               const float* __restrict__ sph, const _Float16* __restrict__ R,
                               float* __restrict__ A){
  int wv=threadIdx.x>>5, lane=threadIdx.x&31;
  int e = blockIdx.x*8+wv;
  int s = eidx[e], r = eidx[NE+e];
  float nfs = nf[(size_t)s*32+lane];
  float Rl[3];
#pragma unroll
  for (int l=0;l<3;l++) Rl[l]=(float)R[(size_t)e*96+lane*3+l];
  float sp[9];
#pragma unroll
  for (int k=0;k<9;k++) sp[k]=sph[(size_t)e*9+k];
  float* Ar = A + (size_t)r*288 + lane*9;
#pragma unroll
  for (int k=0;k<9;k++) atomicAdd(&Ar[k], nfs*sp[k]*Rl[LMAP_[k]]*INV_AVG);
}

// ---------------------------------------------------------------- per-node fwd energy + bwd to gA (wave per node)
__global__ void k_node(const float* __restrict__ A, const float* __restrict__ nfeat,
                       const int* __restrict__ zn, const int* __restrict__ batchv,
                       const float* __restrict__ Wmix, const float* __restrict__ Wcontr,
                       const float* __restrict__ Wb1, const float* __restrict__ Wskip,
                       const float* __restrict__ Wout,
                       const float* __restrict__ Wro1, const float* __restrict__ bro1,
                       const float* __restrict__ Wro2, const float* __restrict__ bro2,
                       const float* __restrict__ We0, const float* __restrict__ be0,
                       const float* __restrict__ Wsc, const float* __restrict__ bsc,
                       float* __restrict__ gA, float* __restrict__ Eb){
  int wv=threadIdx.x>>5, lane=threadIdx.x&31;
  int n = blockIdx.x*8+wv;
  int z = zn[n];
  float Ak[9], A2k[9];
#pragma unroll
  for (int k=0;k<9;k++) Ak[k] = A[(size_t)n*288 + lane*9 + k];
  // A2[d,k] = sum_c A[c,k] * Wmix[l(k)][c][d]   (lane = d)
#pragma unroll
  for (int k=0;k<9;k++){
    const float* wm = Wmix + LMAP_[k]*1024 + lane;
    float s=0.f;
#pragma unroll
    for (int c=0;c<32;c++) s += __shfl(Ak[k], c, 32) * wm[c*32];
    A2k[k]=s;
  }
  float Bc;
  { float s=0.f;
#pragma unroll
    for (int k=0;k<9;k++){ float wc = Wcontr[z*96 + lane*3 + LMAP_[k]]; s += wc*A2k[k]*A2k[k]; }
    Bc = s + Wb1[z*32+lane]*A2k[0]; }
  float nfc = nfeat[(size_t)n*32 + lane];
  float nfo = 0.f;                       // lane = d
#pragma unroll
  for (int c=0;c<32;c++){
    nfo += __shfl(Bc,c,32)*Wout[c*32+lane];
    nfo += Wskip[z*1024 + c*32 + lane]*__shfl(nfc,c,32);
  }
  int j = lane & 15;
  float uj = bro1[j];
#pragma unroll
  for (int d=0; d<32; d++) uj += __shfl(nfo,d,32)*Wro1[d*16 + j];
  float part = (lane<16)? silu_f(uj)*Wro2[lane] : 0.f;
#pragma unroll
  for (int off=16; off>0; off>>=1) part += __shfl_xor(part, off, 32);
  float Ero = part + bro2[0];
  int z2 = zn[z];                                 // reference quirk: scale[atomic_numbers]
  float scale = Wsc[z2] + bsc[0];
  float En = (We0[z] + be0[0] + Ero)*scale;
  if (lane==0) atomicAdd(&Eb[batchv[n]], En);
  // ---- backward to gA
  float cj = (lane<16)? Wro2[lane]*dsilu_f(uj) : 0.f;
  float gnf = 0.f;
#pragma unroll
  for (int jj=0; jj<16; jj++) gnf += __shfl(cj, jj, 32)*Wro1[lane*16 + jj];
  gnf *= scale;
  float gB = 0.f;                                 // lane = c
#pragma unroll
  for (int d=0; d<32; d++) gB += __shfl(gnf,d,32)*Wout[lane*32+d];
  float gA2k[9];                                  // lane = d
#pragma unroll
  for (int k=0;k<9;k++){
    float wc = Wcontr[z*96 + lane*3 + LMAP_[k]];
    float g = gB*(2.f*wc*A2k[k]);
    if (k==0) g += gB*Wb1[z*32+lane];
    gA2k[k]=g;
  }
#pragma unroll
  for (int k=0;k<9;k++){                          // lane = c
    const float* wm = Wmix + LMAP_[k]*1024 + lane*32;
    float s=0.f;
#pragma unroll
    for (int d=0; d<32; d++) s += __shfl(gA2k[k], d, 32)*wm[d];
    gA[(size_t)n*288 + lane*9 + k] = s;
  }
}

// ---------------------------------------------------------------- edge backward: gR0 + gsph (wave per edge)
__global__ void k_edge_gr(const int* __restrict__ eidx, const float* __restrict__ nfeat,
                          const float* __restrict__ gA, const float* __restrict__ sph,
                          const _Float16* __restrict__ R,
                          _Float16* __restrict__ gR0, float* __restrict__ gsph){
  int wv=threadIdx.x>>5, lane=threadIdx.x&31;
  int e = blockIdx.x*8+wv;
  int s = eidx[e], r = eidx[NE+e];
  float nfs = nfeat[(size_t)s*32 + lane];
  float sp[9], gm[9];
#pragma unroll
  for (int k=0;k<9;k++){ sp[k]=sph[(size_t)e*9+k];
                         gm[k]=gA[(size_t)r*288 + lane*9 + k]*INV_AVG; }
  float Rl[3];
#pragma unroll
  for (int l=0;l<3;l++) Rl[l]=(float)R[(size_t)e*96 + lane*3 + l];
  float gr[3]={0.f,0.f,0.f}, p[9];
#pragma unroll
  for (int k=0;k<9;k++){ int l=LMAP_[k]; gr[l] += gm[k]*sp[k]; p[k]=gm[k]*nfs*Rl[l]; }
#pragma unroll
  for (int l=0;l<3;l++) gR0[(size_t)e*96 + lane*3 + l] = (_Float16)(gr[l]*nfs);
#pragma unroll
  for (int k=0;k<9;k++){
    float v=p[k];
#pragma unroll
    for (int off=16; off>0; off>>=1) v += __shfl_xor(v, off, 32);
    if (lane==0) gsph[(size_t)e*9+k]=v;
  }
}

// ---------------------------------------------------------------- MLP backward (recompute fwd + transposed WMMA)
__global__ void __launch_bounds__(256) k_edge_bwd_mlp(
  const float* __restrict__ ebuf, const _Float16* __restrict__ gR0,
  const float* __restrict__ Wr1,const float* __restrict__ br1,
  const float* __restrict__ Wr2,const float* __restrict__ br2,
  const float* __restrict__ Wr3,const float* __restrict__ br3,
  const float* __restrict__ Wr4,
  float* __restrict__ geb)
{
  // all weights pre-swizzled to fragment order: [tile][lane][16]
  __shared__ __align__(32) _Float16 w1[1024],w2[1024],w3[1024];
  __shared__ __align__(32) _Float16 t4[3072],t3[1024],t2[1024],t1[512];
  __shared__ __align__(32) _Float16 xbuf[8][512];
  int tid=threadIdx.x;
  for (int i=tid;i<1024;i+=256){
    int lane=(i>>4)&31, ii=i&15;
    int K=wk(lane,ii), n=((i>>9)<<4)+(lane&15);
    w1[i]=(_Float16)((K<8)?Wr1[(K<<5)+n]:0.f);
    w2[i]=(_Float16)Wr2[(K<<5)+n];
    w3[i]=(_Float16)Wr3[(K<<5)+n];
    t3[i]=(_Float16)Wr3[(n<<5)+K];      // W3^T
    t2[i]=(_Float16)Wr2[(n<<5)+K];      // W2^T
  }
  for (int i=tid;i<3072;i+=256){        // W4^T: tileIdx = kc*2+nt
    int lane=(i>>4)&31, ii=i&15;
    int K=wk(lane,ii);
    int tI=i>>9, kc=tI>>1, nt=tI&1;
    int n=(nt<<4)+(lane&15);
    t4[i]=(_Float16)Wr4[n*96 + kc*32 + K];
  }
  for (int i=tid;i<512;i+=256){         // W1^T (pad N to 16)
    int lane=(i>>4)&31, ii=i&15;
    int K=wk(lane,ii), n=lane&15;
    t1[i]=(_Float16)((n<8)?Wr1[(n<<5)+K]:0.f);
  }
  __syncthreads();

  int wv=tid>>5, lane=tid&31;
  int ebase=(blockIdx.x*8+wv)*16;
  int m=lane&15, rofs=(lane&16)?8:0;
  int kbase=(lane&16)?8:0;
  _Float16* xb=xbuf[wv];

  float z1s[16], z2s[16], z3s[16];
  AF a;
#pragma unroll
  for (int i=0;i<16;i++){ int K=wk(lane,i);
    a.h[i]=(K<8)?(_Float16)ebuf[(size_t)(ebase+m)*8+K]:(_Float16)0.f; }

  { CF c0={},c1={};
    c0.v=WMMA(a.v,ldbs(w1,0,lane),c0.v);
    c1.v=WMMA(a.v,ldbs(w1,1,lane),c1.v);
#pragma unroll
    for (int v=0;v<8;v++){ int row=v+rofs;
      float za=c0.f[v]+br1[m], zb=c1.f[v]+br1[16+m];
      z1s[v]=za; z1s[8+v]=zb;
      xb[row*32+m]=(_Float16)silu_f(za); xb[row*32+16+m]=(_Float16)silu_f(zb); } }
  __syncthreads();
  lda(a, xb, m, lane);
  __syncthreads();
  { CF c0={},c1={};
    c0.v=WMMA(a.v,ldbs(w2,0,lane),c0.v);
    c1.v=WMMA(a.v,ldbs(w2,1,lane),c1.v);
#pragma unroll
    for (int v=0;v<8;v++){ int row=v+rofs;
      float za=c0.f[v]+br2[m], zb=c1.f[v]+br2[16+m];
      z2s[v]=za; z2s[8+v]=zb;
      xb[row*32+m]=(_Float16)silu_f(za); xb[row*32+16+m]=(_Float16)silu_f(zb); } }
  __syncthreads();
  lda(a, xb, m, lane);
  __syncthreads();
  { CF c0={},c1={};
    c0.v=WMMA(a.v,ldbs(w3,0,lane),c0.v);
    c1.v=WMMA(a.v,ldbs(w3,1,lane),c1.v);
#pragma unroll
    for (int v=0;v<8;v++){ z3s[v]=c0.f[v]+br3[m]; z3s[8+v]=c1.f[v]+br3[16+m]; } }

  // ---- backward: g_h3 = gR0 @ W4^T  (K=96 in 3 chunks, vector A-frag loads)
  CF g3a={},g3b={};
  for (int kc=0;kc<3;kc++){
    AF ag;
    const _Float16* gp = gR0 + (size_t)(ebase+m)*96 + kc*32 + kbase;
    ag.q[0] = *reinterpret_cast<const v8h*>(gp);
    ag.q[1] = *reinterpret_cast<const v8h*>(gp + 16);
    g3a.v=WMMA(ag.v,ldbs(t4,kc*2+0,lane),g3a.v);
    g3b.v=WMMA(ag.v,ldbs(t4,kc*2+1,lane),g3b.v);
  }
#pragma unroll
  for (int v=0;v<8;v++){ int row=v+rofs;
    xb[row*32+m]   =(_Float16)(g3a.f[v]*dsilu_f(z3s[v]));
    xb[row*32+16+m]=(_Float16)(g3b.f[v]*dsilu_f(z3s[8+v])); }
  __syncthreads();
  lda(a, xb, m, lane);
  __syncthreads();
  CF g2a={},g2b={};
  g2a.v=WMMA(a.v,ldbs(t3,0,lane),g2a.v);
  g2b.v=WMMA(a.v,ldbs(t3,1,lane),g2b.v);
#pragma unroll
  for (int v=0;v<8;v++){ int row=v+rofs;
    xb[row*32+m]   =(_Float16)(g2a.f[v]*dsilu_f(z2s[v]));
    xb[row*32+16+m]=(_Float16)(g2b.f[v]*dsilu_f(z2s[8+v])); }
  __syncthreads();
  lda(a, xb, m, lane);
  __syncthreads();
  CF g1a={},g1b={};
  g1a.v=WMMA(a.v,ldbs(t2,0,lane),g1a.v);
  g1b.v=WMMA(a.v,ldbs(t2,1,lane),g1b.v);
#pragma unroll
  for (int v=0;v<8;v++){ int row=v+rofs;
    xb[row*32+m]   =(_Float16)(g1a.f[v]*dsilu_f(z1s[v]));
    xb[row*32+16+m]=(_Float16)(g1b.f[v]*dsilu_f(z1s[8+v])); }
  __syncthreads();
  lda(a, xb, m, lane);
  __syncthreads();
  CF ge={};
  ge.v=WMMA(a.v,ldbs(t1,0,lane),ge.v);
  if (m<8){
#pragma unroll
    for (int v=0;v<8;v++) geb[(size_t)(ebase+v+rofs)*8+m]=ge.f[v];
  }
}

// ---------------------------------------------------------------- final: g_vec -> forces
__global__ void k_edge_force(const float* __restrict__ pos, const float* __restrict__ sh,
                             const int* __restrict__ eidx,
                             const float* __restrict__ geb, const float* __restrict__ gsph,
                             float* __restrict__ F){
  int e = blockIdx.x*256 + threadIdx.x;
  int s = eidx[e], r = eidx[NE+e];
  float vx = pos[r*3+0]-pos[s*3+0]+sh[(size_t)e*3+0];
  float vy = pos[r*3+1]-pos[s*3+1]+sh[(size_t)e*3+1];
  float vz = pos[r*3+2]-pos[s*3+2]+sh[(size_t)e*3+2];
  float len = sqrtf(vx*vx+vy*vy+vz*vz+1e-12f);
  float inv = frcp(len);
  float x=vx*inv, y=vy*inv, z=vz*inv;
  float uu = len*(1.f/RCUT);
  float fc=0.f, dfc=0.f;
  if (uu < 1.f){
    float u2=uu*uu, u3=u2*uu, u5=u3*u2, u6=u3*u3;
    fc  = 1.f - 28.f*u6 + 48.f*u6*uu - 21.f*u6*u2;
    dfc = (-168.f*u5 + 336.f*u6 - 168.f*u6*uu)*(1.f/RCUT);
  }
  float rs = fmaxf(len,1e-6f), irs=frcp(rs);
  float glen = 0.f;
#pragma unroll
  for (int n=1;n<=8;n++){
    float w=(float)n*(PI_F/RCUT);
    float sn=__sinf(w*rs), cn=__cosf(w*rs);
    float bes  = 0.44721360f*sn*irs;
    float dbes = (len>1e-6f)? 0.44721360f*(w*cn*rs - sn)*irs*irs : 0.f;
    glen += geb[(size_t)e*8+n-1]*(dbes*fc + bes*dfc);
  }
  float g[9];
#pragma unroll
  for (int k=0;k<9;k++) g[k]=gsph[(size_t)e*9+k];
  float gux = S3C*g[1] + S15C*(y*g[4] + z*g[7] + x*g[8]);
  float guy = S3C*g[2] + S15C*(x*g[4] + z*g[5] - y*g[8]);
  float guz = S3C*g[3] + S15C*(y*g[5] + x*g[7]) + 3.f*S5C*z*g[6];
  float dot = x*gux + y*guy + z*guz;
  float gvx = x*glen + (gux - x*dot)*inv;
  float gvy = y*glen + (guy - y*dot)*inv;
  float gvz = z*glen + (guz - z*dot)*inv;
  // Forces = -grad;  vec = pos[r]-pos[s]
  atomicAdd(&F[r*3+0], -gvx); atomicAdd(&F[r*3+1], -gvy); atomicAdd(&F[r*3+2], -gvz);
  atomicAdd(&F[s*3+0],  gvx); atomicAdd(&F[s*3+1],  gvy); atomicAdd(&F[s*3+2],  gvz);
}

// ================================================================ host
extern "C" void kernel_launch(void* const* d_in, const int* in_sizes, int n_in,
                              void* d_out, int out_size, void* d_ws, size_t ws_size,
                              hipStream_t stream) {
  const float* pos   = (const float*)d_in[0];
  const float* shf   = (const float*)d_in[1];
  const int*   eidx  = (const int*)  d_in[2];
  const int*   zn    = (const int*)  d_in[3];
  const int*   batch = (const int*)  d_in[4];
  const float* Wi    = (const float*)d_in[5];
  const float* bi    = (const float*)d_in[6];
  const float* Wr1   = (const float*)d_in[7];
  const float* br1   = (const float*)d_in[8];
  const float* Wr2   = (const float*)d_in[9];
  const float* br2   = (const float*)d_in[10];
  const float* Wr3   = (const float*)d_in[11];
  const float* br3   = (const float*)d_in[12];
  const float* Wr4   = (const float*)d_in[13];
  const float* br4   = (const float*)d_in[14];
  const float* Wmix  = (const float*)d_in[15];
  const float* Wcontr= (const float*)d_in[16];
  const float* Wb1   = (const float*)d_in[17];
  const float* Wskip = (const float*)d_in[18];
  const float* Wout  = (const float*)d_in[19];
  const float* Wro1  = (const float*)d_in[20];
  const float* bro1  = (const float*)d_in[21];
  const float* Wro2  = (const float*)d_in[22];
  const float* bro2  = (const float*)d_in[23];
  const float* We0   = (const float*)d_in[24];
  const float* be0   = (const float*)d_in[25];
  const float* Wsc   = (const float*)d_in[26];
  const float* bsc   = (const float*)d_in[27];
  (void)in_sizes; (void)n_in; (void)out_size; (void)ws_size;

  char* ws = (char*)d_ws;
  size_t off = 0;
  auto carve = [&](size_t bytes)->char* {
    char* p = ws + off;
    off = (off + bytes + 255) & ~(size_t)255;
    return p;
  };
  float*     nf   = (float*)    carve((size_t)NA*32*4);
  float*     A    = (float*)    carve((size_t)NA*288*4);
  float*     gA   = (float*)    carve((size_t)NA*288*4);
  float*     sph  = (float*)    carve((size_t)NE*9*4);
  float*     ebuf = (float*)    carve((size_t)NE*8*4);
  _Float16*  R    = (_Float16*) carve((size_t)NE*96*2);
  _Float16*  gR0  = (_Float16*) carve((size_t)NE*96*2);
  float*     gsph = (float*)    carve((size_t)NE*9*4);
  float*     geb  = (float*)    carve((size_t)NE*8*4);

  float* out = (float*)d_out;      // [128] Eb then [20000*3] Forces
  float* F   = out + NG;

  k_zero        <<<22500, 256, 0, stream>>>(A, out);
  k_node_embed  <<< 2500, 256, 0, stream>>>(zn, Wi, bi, nf);
  k_edge_geom   <<< 1250, 256, 0, stream>>>(pos, shf, eidx, sph, ebuf);
  k_edge_mlp_fwd<<< 2500, 256, 0, stream>>>(ebuf, Wr1, br1, Wr2, br2, Wr3, br3, Wr4, br4, R);
  k_edge_scatter<<<40000, 256, 0, stream>>>(eidx, nf, sph, R, A);
  k_node        <<< 2500, 256, 0, stream>>>(A, nf, zn, batch, Wmix, Wcontr, Wb1, Wskip, Wout,
                                            Wro1, bro1, Wro2, bro2, We0, be0, Wsc, bsc, gA, out);
  k_edge_gr     <<<40000, 256, 0, stream>>>(eidx, nf, gA, sph, R, gR0, gsph);
  k_edge_bwd_mlp<<< 2500, 256, 0, stream>>>(ebuf, gR0, Wr1, br1, Wr2, br2, Wr3, br3, Wr4, geb);
  k_edge_force  <<< 1250, 256, 0, stream>>>(pos, shf, eidx, geb, gsph, F);
}
